// MultiModalFusion_34291018891543
// MI455X (gfx1250) — matrix operand order repaired and checked by reading, not compiled, not run on hardware
//
#include <hip/hip_runtime.h>
#include <hip/hip_bf16.h>
#include <math.h>
#include <stdint.h>

// ---------------------------------------------------------------------------
// Static problem shape (from reference setup_inputs): B=8, V=T=H=1024, L=2048
// ---------------------------------------------------------------------------
#define BDIM 8
#define NSEQ 1024        // V == T == H == 1024
#define LCAT 2048        // V + T

typedef __attribute__((ext_vector_type(16))) __bf16 v16bf;
typedef __attribute__((ext_vector_type(8)))  __bf16 v8bf;
typedef __attribute__((ext_vector_type(8)))  float  v8f;

#define APITCH 40   // 32 k-elems + 8 pad (80B rows: 16B aligned, bank-staggered)
#define BPITCH 40

// Concatenate two v8bf into one v16bf fragment
__device__ __forceinline__ v16bf cat8(v8bf a, v8bf b) {
  return __builtin_shufflevector(a, b, 0,1,2,3,4,5,6,7,8,9,10,11,12,13,14,15);
}

// ---------------------------------------------------------------------------
// CDNA5 async global->LDS copy (tracked by ASYNCcnt, 16B per lane).
// LDS operand is the low 32 bits of the generic pointer (LDS aperture offset).
// ---------------------------------------------------------------------------
__device__ __forceinline__ void async_ld_b128(void* lds, const void* gaddr) {
  const unsigned loff = (unsigned)(unsigned long long)(uintptr_t)lds;
  asm volatile("global_load_async_to_lds_b128 %0, %1, off"
               :: "v"(loff), "v"(gaddr) : "memory");
}
__device__ __forceinline__ void wait_asynccnt0() {
  asm volatile("s_wait_asynccnt 0" ::: "memory");
}

// Stage a 128x32 bf16 tile (row-major src, ld elems) into LDS rows of APITCH
// using async b128 transfers: 512 16B chunks, 2 per thread.
__device__ __forceinline__ void stage_async_bf16(__bf16* ldsTile,
                                                 const __bf16* src, long ld,
                                                 int r0, int k0, int tid) {
#pragma unroll
  for (int it = 0; it < 2; ++it) {
    const int c2 = tid * 2 + it;        // 0..511
    const int row = c2 >> 2;            // 0..127
    const int ce = (c2 & 3) * 8;        // element offset: 0,8,16,24
    const __bf16* g = src + (long)(r0 + row) * ld + k0 + ce;
    async_ld_b128((void*)(ldsTile + row * APITCH + ce), (const void*)g);
  }
}

// ---------------------------------------------------------------------------
// WMMA core: one BK=32 step of a 128x128 tile.
// 8 waves in a 2(M) x 4(N) grid; each wave owns a 64x32 region = 4x2 WMMA tiles.
// A fragment layout (16-bit A 16x32, ISA 7.12.2): lanes 0-15 row m, halves
// k={0..7,16..23}; lanes 16-31 same rows, k={8..15,24..31}.
// B tile staged n-major in LDS ([n][k]); B fragment uses the same k order as A:
// lanes 0-15 k={0..7,16..23}; lanes 16-31 k={8..15,24..31}.
// ---------------------------------------------------------------------------
__device__ __forceinline__ void mma_step(const __bf16* As, const __bf16* Bs,
                                         v8f acc[4][2], int wm, int wn, int lane)
{
  const int lrow = lane & 15;
  const int hsel = (lane >> 4) & 1;
  v16bf a[4], bfr[2];
#pragma unroll
  for (int mt = 0; mt < 4; ++mt) {
    const __bf16* p = As + (wm * 64 + mt * 16 + lrow) * APITCH + hsel * 8;
    a[mt] = cat8(*(const v8bf*)p, *(const v8bf*)(p + 16));
  }
#pragma unroll
  for (int nt = 0; nt < 2; ++nt) {
    const __bf16* p = Bs + (wn * 32 + nt * 16 + lrow) * BPITCH + hsel * 8;
    bfr[nt] = cat8(*(const v8bf*)p, *(const v8bf*)(p + 16));
  }
#pragma unroll
  for (int mt = 0; mt < 4; ++mt)
#pragma unroll
    for (int nt = 0; nt < 2; ++nt)
      acc[mt][nt] = __builtin_amdgcn_wmma_f32_16x16x32_bf16(
          false, a[mt], false, bfr[nt], (short)0, acc[mt][nt], false, false);
}

// Accumulator element -> (m,n) mapping within the 128x128 tile (ISA C/D layout):
// VGPR r, lanes 0-15 -> row r, lanes 16-31 -> row r+8; col = lane & 15.
template <typename F>
__device__ __forceinline__ void epilogue_for(v8f acc[4][2], int wm, int wn,
                                             int lane, F f)
{
  const int lr = lane & 15;
  const int lh = (lane & 16) ? 8 : 0;
#pragma unroll
  for (int mt = 0; mt < 4; ++mt)
#pragma unroll
    for (int nt = 0; nt < 2; ++nt)
#pragma unroll
      for (int r = 0; r < 8; ++r)
        f(wm * 64 + mt * 16 + r + lh, wn * 32 + nt * 16 + lr, acc[mt][nt][r]);
}

// ---------------------------------------------------------------------------
// LDS staging helpers (256 threads; 128x32 tiles; convert fp32->bf16 inline)
// ---------------------------------------------------------------------------
__device__ __forceinline__ void stageA_f32(__bf16* As, const float* A, long ld,
                                           int m0, int k0, int tid) {
  const int row = tid >> 1, c = (tid & 1) * 16;
  const float* src = A + (long)(m0 + row) * ld + k0 + c;
  __builtin_prefetch(src + 32, 0, 1);   // next k-tile -> global_prefetch_b8
  __bf16* dst = As + row * APITCH + c;
#pragma unroll
  for (int i = 0; i < 16; ++i) dst[i] = (__bf16)src[i];
}

// src is row-major [n][k] (e.g. torch Linear weight W[o][h]); direct copy.
__device__ __forceinline__ void stageB_nk_f32(__bf16* Bs, const float* src, long ld,
                                              int n0, int k0, int tid) {
  const int n = tid >> 1, c = (tid & 1) * 16;
  const float* s = src + (long)(n0 + n) * ld + k0 + c;
  __builtin_prefetch(s + 32, 0, 1);
  __bf16* d = Bs + n * BPITCH + c;
#pragma unroll
  for (int i = 0; i < 16; ++i) d[i] = (__bf16)s[i];
}

// src is row-major [k][n]; transpose into LDS [n][k]. Optional column bound
// (shift/colmax) implements the conv zero-padding at the sequence edges.
__device__ __forceinline__ void stageB_kn_bf16(__bf16* Bs, const __bf16* src, long ld,
                                               int n0, int shift, int colmax, int tid) {
  const int k = tid & 31;
  const int nseg = tid >> 5;        // 0..7
  const __bf16* s = src + (long)k * ld;
#pragma unroll
  for (int i = 0; i < 16; ++i) {
    const int n = nseg * 16 + i;
    const int col = n0 + n + shift;
    __bf16 v = (colmax < 0 || (unsigned)col < (unsigned)colmax) ? s[col] : (__bf16)0.0f;
    Bs[n * BPITCH + k] = v;
  }
}

#define GEMM_PROLOGUE                                                          \
  __shared__ __attribute__((aligned(16))) __bf16 As[128 * APITCH];             \
  __shared__ __attribute__((aligned(16))) __bf16 Bs[128 * BPITCH];             \
  const int tid = threadIdx.x;                                                 \
  const int lane = tid & 31;                                                   \
  const int wave = tid >> 5;                                                   \
  const int wm = wave >> 2;                                                    \
  const int wn = wave & 3;                                                     \
  v8f acc[4][2];                                                               \
  _Pragma("unroll") for (int _i = 0; _i < 4; ++_i)                             \
  _Pragma("unroll") for (int _j = 0; _j < 2; ++_j) acc[_i][_j] = (v8f)0.0f;

// ---------------------------------------------------------------------------
// K1: the six projection GEMMs  Y = X @ W^T + bias  (fp32 in, bf16 out)
// ---------------------------------------------------------------------------
__global__ __launch_bounds__(256) void proj_gemm_kernel(
    const float* __restrict__ visual, const float* __restrict__ text,
    const float* __restrict__ Wq, const float* __restrict__ bq,
    const float* __restrict__ Wk, const float* __restrict__ bk,
    const float* __restrict__ Wv, const float* __restrict__ bv,
    __bf16* Qt, __bf16* Kv, __bf16* Vv, __bf16* Qv, __bf16* Kt, __bf16* Vt)
{
  GEMM_PROLOGUE
  const int m0 = blockIdx.x * 128;
  const int n0 = blockIdx.y * 128;
  const float* X; const float* W; const float* bias; __bf16* dst;
  switch (blockIdx.z) {
    case 0:  X = text;   W = Wq; bias = bq; dst = Qt; break;
    case 1:  X = visual; W = Wk; bias = bk; dst = Kv; break;
    case 2:  X = visual; W = Wv; bias = bv; dst = Vv; break;
    case 3:  X = visual; W = Wq; bias = bq; dst = Qv; break;
    case 4:  X = text;   W = Wk; bias = bk; dst = Kt; break;
    default: X = text;   W = Wv; bias = bv; dst = Vt; break;
  }
  for (int k0 = 0; k0 < NSEQ; k0 += 32) {
    __syncthreads();
    stageA_f32(As, X, NSEQ, m0, k0, tid);
    stageB_nk_f32(Bs, W, NSEQ, n0, k0, tid);   // B[k=h][n=o] = W[o][h]
    __syncthreads();
    mma_step(As, Bs, acc, wm, wn, lane);
  }
  epilogue_for(acc, wm, wn, lane, [&](int ml, int nl, float v) {
    const int m = m0 + ml;
    const int n = n0 + nl;
    dst[(size_t)m * NSEQ + n] = (__bf16)(v + bias[n]);
  });
}

// ---------------------------------------------------------------------------
// K2: scores S = Q K^T with analytic chart-sparse mask (-inf where masked)
// Q and K tiles staged via GLOBAL_LOAD_ASYNC_TO_LDS_B128 (ASYNCcnt).
// dir 0 = text->visual (q=text, k=visual), dir 1 = visual->text
// ---------------------------------------------------------------------------
__global__ __launch_bounds__(256) void attn_scores_kernel(
    const __bf16* __restrict__ Qt, const __bf16* __restrict__ Qv,
    const __bf16* __restrict__ Kv, const __bf16* __restrict__ Kt,
    float* __restrict__ S)
{
  GEMM_PROLOGUE
  const int m0 = blockIdx.x * 128;
  const int n0 = blockIdx.y * 128;
  const int zi = blockIdx.z;
  const int dir = zi >> 3;
  const int b = zi & 7;
  const __bf16* Q = (dir == 0 ? Qt : Qv) + (size_t)b * NSEQ * NSEQ;
  const __bf16* K = (dir == 0 ? Kv : Kt) + (size_t)b * NSEQ * NSEQ;
  float* Sp = S + (size_t)zi * NSEQ * NSEQ;

  for (int k0 = 0; k0 < NSEQ; k0 += 32) {
    __syncthreads();
    stage_async_bf16(As, Q, NSEQ, m0, k0, tid);          // A[m][k] = Q[q][h]
    stage_async_bf16(Bs, K, NSEQ, n0, k0, tid);          // B[n][k] = K[key][h]
    wait_asynccnt0();
    __syncthreads();
    mma_step(As, Bs, acc, wm, wn, lane);
  }
  epilogue_for(acc, wm, wn, lane, [&](int ml, int nl, float v) {
    const int q = m0 + ml;
    const int kk = n0 + nl;
    bool msk;
    if (dir == 0) {                // t2v: [T,V]
      msk = (q < 5) || (q >= 341 && q < 682 && kk >= 256 && kk < 768);
    } else {                       // v2t: [V,T], window width 8
      const int st = (q < 1016) ? q : 1016;
      msk = (kk < 5) || (kk >= st && kk < st + 8);
    }
    Sp[(size_t)q * NSEQ + kk] = msk ? v : -__builtin_inff();
  });
}

// ---------------------------------------------------------------------------
// K3: row softmax over 1024 keys; fully-masked rows -> zeros; bf16 output P
// ---------------------------------------------------------------------------
__global__ __launch_bounds__(256) void softmax_kernel(const float* __restrict__ S,
                                                      __bf16* __restrict__ P)
{
  __shared__ float redm[8];
  __shared__ float reds[8];
  const size_t row = blockIdx.x;
  const float* s = S + row * NSEQ;
  __bf16* p = P + row * NSEQ;
  const int tid = threadIdx.x;
  const int lane = tid & 31;
  const int wave = tid >> 5;

  float v[4];
  float mx = -__builtin_inff();
#pragma unroll
  for (int i = 0; i < 4; ++i) { v[i] = s[tid + 256 * i]; mx = fmaxf(mx, v[i]); }
#pragma unroll
  for (int o = 16; o > 0; o >>= 1) mx = fmaxf(mx, __shfl_xor(mx, o, 32));
  if (lane == 0) redm[wave] = mx;
  __syncthreads();
  if (wave == 0) {
    float m2 = (lane < 8) ? redm[lane] : -__builtin_inff();
#pragma unroll
    for (int o = 4; o > 0; o >>= 1) m2 = fmaxf(m2, __shfl_xor(m2, o, 32));
    if (lane == 0) redm[0] = m2;
  }
  __syncthreads();
  mx = redm[0];

  const bool valid = (mx > -__builtin_inff());
  float e[4];
  float sum = 0.f;
#pragma unroll
  for (int i = 0; i < 4; ++i) { e[i] = valid ? __expf(v[i] - mx) : 0.f; sum += e[i]; }
#pragma unroll
  for (int o = 16; o > 0; o >>= 1) sum += __shfl_xor(sum, o, 32);
  if (lane == 0) reds[wave] = sum;
  __syncthreads();
  if (wave == 0) {
    float s2 = (lane < 8) ? reds[lane] : 0.f;
#pragma unroll
    for (int o = 4; o > 0; o >>= 1) s2 += __shfl_xor(s2, o, 32);
    if (lane == 0) reds[0] = s2;
  }
  __syncthreads();
  const float tot = reds[0];
  const float inv = (tot > 0.f) ? 1.0f / tot : 0.f;
#pragma unroll
  for (int i = 0; i < 4; ++i) p[tid + 256 * i] = (__bf16)(e[i] * inv);
}

// ---------------------------------------------------------------------------
// K4: O = P @ V + residual, stored transposed into xcat[B, H, L] (bf16)
// P tile staged async; V tile needs a transpose so it stays on the VALU path.
// dir 0 (t2v) -> text rows (L offset V), dir 1 (v2t) -> visual rows (L offset 0)
// ---------------------------------------------------------------------------
__global__ __launch_bounds__(256) void attn_out_kernel(
    const __bf16* __restrict__ P,
    const __bf16* __restrict__ Vv, const __bf16* __restrict__ Vt,
    const float* __restrict__ visual, const float* __restrict__ text,
    __bf16* __restrict__ xcat)
{
  GEMM_PROLOGUE
  const int m0 = blockIdx.x * 128;
  const int n0 = blockIdx.y * 128;
  const int zi = blockIdx.z;
  const int dir = zi >> 3;
  const int b = zi & 7;
  const __bf16* A  = P + (size_t)zi * NSEQ * NSEQ;
  const __bf16* Vm = (dir == 0 ? Vv : Vt) + (size_t)b * NSEQ * NSEQ;
  const float* res = (dir == 0 ? text : visual) + (size_t)b * NSEQ * NSEQ;
  const int lbase = (dir == 0) ? NSEQ : 0;

  for (int k0 = 0; k0 < NSEQ; k0 += 32) {
    __syncthreads();
    stage_async_bf16(As, A, NSEQ, m0, k0, tid);                       // A[q][key]
    stageB_kn_bf16(Bs, Vm + (size_t)k0 * NSEQ, NSEQ, n0, 0, -1, tid); // B[k=key][n=h]
    wait_asynccnt0();
    __syncthreads();
    mma_step(As, Bs, acc, wm, wn, lane);
  }
  epilogue_for(acc, wm, wn, lane, [&](int ml, int nl, float v) {
    const int q = m0 + ml;
    const int h = n0 + nl;
    const float outv = v + res[(size_t)q * NSEQ + h];
    xcat[((size_t)b * NSEQ + h) * LCAT + lbase + q] = (__bf16)outv;
  });
}

// ---------------------------------------------------------------------------
// K5: grouped conv1 (groups=8, k=3, pad=1) as 3-tap shifted GEMM + exact GELU
// out tile: 128 out-channels (one group) x 128 positions
// ---------------------------------------------------------------------------
__global__ __launch_bounds__(256) void conv1_gelu_kernel(
    const __bf16* __restrict__ xcat, const float* __restrict__ Wc1,
    const float* __restrict__ bc1, __bf16* __restrict__ y1)
{
  GEMM_PROLOGUE
  const int l0 = blockIdx.x * 128;
  const int g  = blockIdx.y;
  const int b  = blockIdx.z;
  const __bf16* xb = xcat + (size_t)b * NSEQ * LCAT;

  for (int dk = 0; dk < 3; ++dk) {
    for (int kt = 0; kt < 4; ++kt) {
      __syncthreads();
      { // A[m][k] = Wc1[g*128+m][kt*32+k][dk]   (fp32, innermost stride 3)
        const int row = tid >> 1;
        const int c = (tid & 1) * 16;
        const float* wsrc = Wc1 + ((size_t)(g * 128 + row) * 128 + kt * 32 + c) * 3 + dk;
        __bf16* dstA = As + row * APITCH + c;
#pragma unroll
        for (int i = 0; i < 16; ++i) dstA[i] = (__bf16)wsrc[i * 3];
      }
      stageB_kn_bf16(Bs, xb + (size_t)(g * 128 + kt * 32) * LCAT, LCAT,
                     l0, dk - 1, LCAT, tid);   // zero-pad at sequence edges
      __syncthreads();
      mma_step(As, Bs, acc, wm, wn, lane);
    }
  }
  epilogue_for(acc, wm, wn, lane, [&](int ml, int nl, float v) {
    const int oc = g * 128 + ml;
    const int pos = l0 + nl;
    const float x = v + bc1[oc];
    const float gl = 0.5f * x * (1.0f + erff(x * 0.70710678118654752f)); // exact GELU
    y1[((size_t)b * NSEQ + oc) * LCAT + pos] = (__bf16)gl;
  });
}

// ---------------------------------------------------------------------------
// K6: pointwise conv2 (1024->1024) + bias, fp32 output transposed to [B,L,H]
// ---------------------------------------------------------------------------
__global__ __launch_bounds__(256) void conv2_kernel(
    const __bf16* __restrict__ y1, const float* __restrict__ Wc2,
    const float* __restrict__ bc2, float* __restrict__ out)
{
  GEMM_PROLOGUE
  const int l0 = blockIdx.x * 128;
  const int o0 = blockIdx.y * 128;
  const int b  = blockIdx.z;
  const __bf16* yb = y1 + (size_t)b * NSEQ * LCAT;

  for (int k0 = 0; k0 < NSEQ; k0 += 32) {
    __syncthreads();
    stageA_f32(As, Wc2, NSEQ, o0, k0, tid);                       // W2[o][i]
    stageB_kn_bf16(Bs, yb + (size_t)k0 * LCAT, LCAT, l0, 0, -1, tid); // B[k=ic][n=l]
    __syncthreads();
    mma_step(As, Bs, acc, wm, wn, lane);
  }
  epilogue_for(acc, wm, wn, lane, [&](int ml, int nl, float v) {
    const int oc = o0 + ml;
    const int pos = l0 + nl;
    out[((size_t)b * LCAT + pos) * NSEQ + oc] = v + bc2[oc];
  });
}

// ---------------------------------------------------------------------------
// Launch: 6-stage pipeline. Workspace layout (bytes):
//   [0,96M)    : Qt,Kv,Vv,Qv,Kt,Vt  (bf16, 16 MiB each)  -- dead after K4
//   [96M,160M) : scores S           (fp32, 64 MiB)
//   [160M,192M): probs  P           (bf16, 32 MiB)
//   [192M,224M): xcat [B,H,L]       (bf16, 32 MiB)
//   [0,32M)    : y1   [B,H,L]       (bf16, overlaid on dead proj region)
// Peak: 224 MiB.
// ---------------------------------------------------------------------------
extern "C" void kernel_launch(void* const* d_in, const int* in_sizes, int n_in,
                              void* d_out, int out_size, void* d_ws, size_t ws_size,
                              hipStream_t stream) {
  (void)in_sizes; (void)n_in; (void)out_size; (void)ws_size;
  const float* visual = (const float*)d_in[0];
  const float* text   = (const float*)d_in[1];
  const float* Wq  = (const float*)d_in[2];
  const float* bq  = (const float*)d_in[3];
  const float* Wk  = (const float*)d_in[4];
  const float* bk  = (const float*)d_in[5];
  const float* Wv  = (const float*)d_in[6];
  const float* bv  = (const float*)d_in[7];
  const float* Wc1 = (const float*)d_in[8];
  const float* bc1 = (const float*)d_in[9];
  const float* Wc2 = (const float*)d_in[10];
  const float* bc2 = (const float*)d_in[11];
  float* out = (float*)d_out;

  char* wsb = (char*)d_ws;
  const size_t PROJ_BYTES = (size_t)BDIM * NSEQ * NSEQ * sizeof(__bf16); // 16 MiB
  __bf16* Qt = (__bf16*)(wsb + 0 * PROJ_BYTES);
  __bf16* Kv = (__bf16*)(wsb + 1 * PROJ_BYTES);
  __bf16* Vv = (__bf16*)(wsb + 2 * PROJ_BYTES);
  __bf16* Qv = (__bf16*)(wsb + 3 * PROJ_BYTES);
  __bf16* Kt = (__bf16*)(wsb + 4 * PROJ_BYTES);
  __bf16* Vt = (__bf16*)(wsb + 5 * PROJ_BYTES);
  float*  S    = (float*)(wsb + 6 * PROJ_BYTES);
  __bf16* P    = (__bf16*)(wsb + 6 * PROJ_BYTES + (size_t)2 * BDIM * NSEQ * NSEQ * sizeof(float));
  __bf16* xcat = (__bf16*)((char*)P + (size_t)2 * BDIM * NSEQ * NSEQ * sizeof(__bf16));
  __bf16* y1   = (__bf16*)wsb;  // overlay: proj region is dead once K4 is done

  const dim3 blk(256);
  proj_gemm_kernel<<<dim3(64, 8, 6), blk, 0, stream>>>(
      visual, text, Wq, bq, Wk, bk, Wv, bv, Qt, Kv, Vv, Qv, Kt, Vt);
  attn_scores_kernel<<<dim3(8, 8, 16), blk, 0, stream>>>(Qt, Qv, Kv, Kt, S);
  softmax_kernel<<<dim3(2 * BDIM * NSEQ), blk, 0, stream>>>(S, P);
  attn_out_kernel<<<dim3(8, 8, 16), blk, 0, stream>>>(P, Vv, Vt, visual, text, xcat);
  conv1_gelu_kernel<<<dim3(16, 8, 8), blk, 0, stream>>>(xcat, Wc1, bc1, y1);
  conv2_kernel<<<dim3(16, 8, 8), blk, 0, stream>>>(y1, Wc2, bc2, out);
}
